// LPAPBottleneckBranch_22754736734235
// MI455X (gfx1250) — compile-verified, matrix operand order checked by reading, not run on hardware
//
#include <hip/hip_runtime.h>
#include <hip/hip_bf16.h>
#include <math.h>

// ---------------- problem constants ----------------
#define BB 8
#define NN 8192
#define CC 256
#define STRIDE 32
#define DD 256
#define HH 8
#define HD 32          // D/H
#define LL 2
#define KEFF 22        // int(4*log(256))
#define TOK (BB*CC)    // 2048 token rows

typedef __attribute__((ext_vector_type(16))) __bf16 v16bf;
typedef __attribute__((ext_vector_type(8)))  __bf16 bf16x8;
typedef __attribute__((ext_vector_type(8)))  float  v8f;

// ---------------- device helpers ----------------
__device__ __forceinline__ float gelu_f(float x) {
    const float k = 0.7978845608028654f;
    float y = k * (x + 0.044715f * x * x * x);
    // branch-free tanh via v_exp_f32
    float t = __expf(-2.0f * fabsf(y));
    float th = (1.0f - t) / (1.0f + t);
    th = copysignf(th, y);
    return 0.5f * x * (1.0f + th);
}

__device__ __forceinline__ v16bf pack16(float4 x0, float4 x1, float4 x2, float4 x3) {
    v16bf r;
    r[0]  = (__bf16)x0.x; r[1]  = (__bf16)x0.y; r[2]  = (__bf16)x0.z; r[3]  = (__bf16)x0.w;
    r[4]  = (__bf16)x1.x; r[5]  = (__bf16)x1.y; r[6]  = (__bf16)x1.z; r[7]  = (__bf16)x1.w;
    r[8]  = (__bf16)x2.x; r[9]  = (__bf16)x2.y; r[10] = (__bf16)x2.z; r[11] = (__bf16)x2.w;
    r[12] = (__bf16)x3.x; r[13] = (__bf16)x3.y; r[14] = (__bf16)x3.z; r[15] = (__bf16)x3.w;
    return r;
}

// f32 source, two contiguous 8-float runs: [0,8) and [16,24) relative to p
__device__ __forceinline__ v16bf load_frag_f32(const float* p) {
    float4 x0 = *(const float4*)(p);
    float4 x1 = *(const float4*)(p + 4);
    float4 x2 = *(const float4*)(p + 16);
    float4 x3 = *(const float4*)(p + 20);
    return pack16(x0, x1, x2, x3);
}

// bf16 source, two contiguous 8-elem (16B) runs: [0,8) and [16,24) relative to p
__device__ __forceinline__ v16bf load_frag_bf16(const __bf16* p) {
    bf16x8 lo = *(const bf16x8*)(p);
    bf16x8 hi = *(const bf16x8*)(p + 16);
    v16bf r;
#pragma unroll
    for (int e = 0; e < 8; ++e) { r[e] = lo[e]; r[e + 8] = hi[e]; }
    return r;
}

__device__ __forceinline__ float blk_sum(float v, float* sh) {
    int t = threadIdx.x;
    sh[t] = v; __syncthreads();
    for (int s = 128; s > 0; s >>= 1) { if (t < s) sh[t] += sh[t + s]; __syncthreads(); }
    float r = sh[0]; __syncthreads();
    return r;
}
__device__ __forceinline__ float blk_max(float v, float* sh) {
    int t = threadIdx.x;
    sh[t] = v; __syncthreads();
    for (int s = 128; s > 0; s >>= 1) { if (t < s) sh[t] = fmaxf(sh[t], sh[t + s]); __syncthreads(); }
    float r = sh[0]; __syncthreads();
    return r;
}

// ---------------- generic WMMA bf16 GEMM (K % 32 == 0) ----------------
// C = act(scale * A@B + bias) + residual ; batched via grid.y/.z with strides
// Each wave computes a 16(M) x 32(N) tile with two accumulators.
// BMODE: 0 = B f32 row-major (strided), 1 = B f32 col-major contiguous,
//        2 = B bf16 col-major contiguous (pre-transposed weights)
struct Gemm {
    const float*  A; long lda, sAy, sAz;
    const float*  B; long ldb, sBy, sBz; int bmode;
    const __bf16* Bbf;
    const float*  bias;
    const float*  residual;
    float* C; long ldc, sCy, sCz;
    int M, K, N; float scale; int act;  // act: 0=none, 1=gelu
};

template <int BMODE>
__global__ __launch_bounds__(128) void gemm_wmma_kernel(Gemm g) {
    int wave = threadIdx.x >> 5;
    int lane = threadIdx.x & 31;
    int tilesN = g.N >> 5;                  // 32-wide tiles
    long tile = (long)blockIdx.x * 4 + wave;
    if (tile >= (long)(g.M >> 4) * tilesN) return;   // wave-uniform exit
    int tm = (int)(tile / tilesN);
    int tn = (int)(tile % tilesN);

    const float* A = g.A + (long)blockIdx.y * g.sAy + (long)blockIdx.z * g.sAz;
    const float* B = g.B + (long)blockIdx.y * g.sBy + (long)blockIdx.z * g.sBz;
    float*       C = g.C + (long)blockIdx.y * g.sCy + (long)blockIdx.z * g.sCz;
    const float* R = g.residual
        ? g.residual + (long)blockIdx.y * g.sCy + (long)blockIdx.z * g.sCz : nullptr;

    int half = lane >> 4;
    int idx  = lane & 15;
    long row  = tm * 16 + idx;              // A row held by this lane
    int  col0 = tn * 32 + idx;              // first B/C column held by this lane
    int  col1 = col0 + 16;

    v8f acc0 = {0.f, 0.f, 0.f, 0.f, 0.f, 0.f, 0.f, 0.f};
    v8f acc1 = {0.f, 0.f, 0.f, 0.f, 0.f, 0.f, 0.f, 0.f};

    const float*  Arow = A + row * g.lda + half * 8;
    const __bf16* Wt0  = g.Bbf + (long)col0 * g.ldb + half * 8;
    const __bf16* Wt1  = g.Bbf + (long)col1 * g.ldb + half * 8;

    for (int k0 = 0; k0 < g.K; k0 += 32) {
        v16bf a = load_frag_f32(Arow + k0);
        v16bf b0, b1;
        if (BMODE == 2) {
            b0 = load_frag_bf16(Wt0 + k0);
            b1 = load_frag_bf16(Wt1 + k0);
        } else if (BMODE == 1) {
            b0 = load_frag_f32(B + (long)col0 * g.ldb + half * 8 + k0);
            b1 = load_frag_f32(B + (long)col1 * g.ldb + half * 8 + k0);
        } else {
            const float* pb  = B + (long)(k0 + half * 8) * g.ldb;
            const float* pb2 = pb + 16 * g.ldb;
#pragma unroll
            for (int e = 0; e < 8; ++e) {
                b0[e]     = (__bf16)pb [(long)e * g.ldb + col0];
                b1[e]     = (__bf16)pb [(long)e * g.ldb + col1];
                b0[e + 8] = (__bf16)pb2[(long)e * g.ldb + col0];
                b1[e + 8] = (__bf16)pb2[(long)e * g.ldb + col1];
            }
        }
        acc0 = __builtin_amdgcn_wmma_f32_16x16x32_bf16(
            false, a, false, b0, (short)0, acc0, false, false);
        acc1 = __builtin_amdgcn_wmma_f32_16x16x32_bf16(
            false, a, false, b1, (short)0, acc1, false, false);
    }

#pragma unroll
    for (int v = 0; v < 8; ++v) {
        int m = tm * 16 + v + half * 8;
        float v0 = acc0[v] * g.scale;
        float v1 = acc1[v] * g.scale;
        if (g.bias) { v0 += g.bias[col0]; v1 += g.bias[col1]; }
        if (g.act == 1) { v0 = gelu_f(v0); v1 = gelu_f(v1); }
        if (R) {
            v0 += R[(long)m * g.ldc + col0];
            v1 += R[(long)m * g.ldc + col1];
        }
        C[(long)m * g.ldc + col0] = v0;
        C[(long)m * g.ldc + col1] = v1;
    }
}

// transpose + convert weight: W (Korig x N, f32) -> Wt (N x Kpad, bf16), zero-pad K
__global__ void transpose_cvt_kernel(const float* W, __bf16* Wt,
                                     int Korig, int Kpad, int N) {
    long i = (long)blockIdx.x * 256 + threadIdx.x;   // over N*Kpad
    int k = (int)(i % Kpad);
    int n = (int)(i / Kpad);
    Wt[i] = (k < Korig) ? (__bf16)W[(long)k * N + n] : (__bf16)0.f;
}

// ---------------- elementwise / reduction kernels ----------------
__global__ void permute_kernel(const float* e, const int* perm, float* ep) {
    int j = blockIdx.x * 256 + threadIdx.x;
    int b = blockIdx.y;
    ep[(long)b * NN + j] = e[(long)b * NN + perm[j]];
}

__global__ void tokens_kernel(const float* ep, float* tokens) {
    int t = blockIdx.x;            // token row b*256+c
    int s = threadIdx.x;           // 0..31
    int b = t >> 8, c = t & 255;
    tokens[(long)t * STRIDE + s] = ep[(long)b * NN + s * CC + c];
}

// zero-pad decoder tokens (2048x3 -> 2048x32)
__global__ void pad_dec_tokens_kernel(const float* dt, float* padded) {
    int t = blockIdx.x;
    int s = threadIdx.x;           // 0..31
    padded[(long)t * 32 + s] = (s < 3) ? dt[(long)t * 3 + s] : 0.f;
}

__global__ void add_pos_kernel(float* x, const float* pos) {
    long i = (long)blockIdx.x * 256 + threadIdx.x;   // TOK*256 elems
    int d = (int)(i & 255);
    long t = i >> 8;
    int c = (int)(t & 255);
    x[i] += pos[(long)c * DD + d];
}

__global__ void layernorm_kernel(const float* x, const float* g, const float* b, float* y) {
    __shared__ float sh[256];
    long row = blockIdx.x;
    int t = threadIdx.x;
    float v = x[row * DD + t];
    float mean = blk_sum(v, sh) * (1.0f / 256.0f);
    float d = v - mean;
    float var = blk_sum(d * d, sh) * (1.0f / 256.0f);
    float r = rsqrtf(var + 1e-5f);
    y[row * DD + t] = d * r * g[t] + b[t];
}

__global__ void softmax256_kernel(float* x) {
    __shared__ float sh[256];
    long row = blockIdx.x;
    int t = threadIdx.x;
    float v = x[row * 256 + t];
    float mx = blk_max(v, sh);
    float e = __expf(v - mx);
    float s = blk_sum(e, sh);
    x[row * 256 + t] = e / s;
}

// softmax over N=8192 + moments: val, dexp, ent -> dec_tokens
__global__ void sur_post_kernel(const float* logits, const float* ep, float* dec_tokens) {
    __shared__ float sh[256];
    int row = blockIdx.x;          // b*256+c
    int b = row >> 8, c = row & 255;
    const float* L = logits + (long)row * NN;
    const float* E = ep + (long)b * NN;
    int t = threadIdx.x;
    float mx = -INFINITY;
    for (int n = t; n < NN; n += 256) mx = fmaxf(mx, L[n]);
    mx = blk_max(mx, sh);
    float s = 0.f, t1 = 0.f, vn = 0.f, dn = 0.f;
    for (int n = t; n < NN; n += 256) {
        float l = L[n] - mx;
        float e = __expf(l);
        s  += e;
        t1 += e * l;
        vn += e * E[n];
        dn += e * (float)((c - (n & 255) + 256) & 255);
    }
    s  = blk_sum(s, sh);
    t1 = blk_sum(t1, sh);
    vn = blk_sum(vn, sh);
    dn = blk_sum(dn, sh);
    if (t == 0) {
        dec_tokens[(long)row * 3 + 0] = vn / s;
        dec_tokens[(long)row * 3 + 1] = dn / s;
        dec_tokens[(long)row * 3 + 2] = logf(s) - t1 / s;
    }
}

__global__ void tval_kernel(const float* slt, float* outT) {
    float x = slt[0];
    float sp = (x > 20.f) ? x : log1pf(expf(x));
    outT[0] = sp + 1e-4f;
}

// sp = softmax(dec_logits / T), doubt, support
__global__ void dec_post_kernel(const float* logits, const float* slt,
                                float* sp, float* doubt, float* support) {
    __shared__ float sh[256];
    int row = blockIdx.x;
    const float* L = logits + (long)row * NN;
    int t = threadIdx.x;
    float x = slt[0];
    float T = ((x > 20.f) ? x : log1pf(expf(x))) + 1e-4f;
    float invT = 1.0f / T;
    float mx = -INFINITY;
    for (int n = t; n < NN; n += 256) mx = fmaxf(mx, L[n] * invT);
    mx = blk_max(mx, sh);
    float s = 0.f, t1 = 0.f;
    for (int n = t; n < NN; n += 256) {
        float l = L[n] * invT - mx;
        float e = __expf(l);
        s += e; t1 += e * l;
    }
    s  = blk_sum(s, sh);
    t1 = blk_sum(t1, sh);
    float inv_s = 1.0f / s;
    for (int n = t; n < NN; n += 256)
        sp[(long)row * NN + n] = __expf(L[n] * invT - mx) * inv_s;
    if (t == 0) {
        doubt[row]   = (logf(s) - t1 * inv_s) / logf((float)NN);
        support[row] = inv_s;
    }
}

__global__ void projected_kernel(const float* sp, const float* dec_tokens,
                                 const int* perm, float* projected) {
    int b = blockIdx.y;
    int j = blockIdx.x * 256 + threadIdx.x;
    __shared__ float vsh[256];
    vsh[threadIdx.x] = dec_tokens[((long)((b << 8) + threadIdx.x)) * 3];
    __syncthreads();
    float acc = 0.f;
    const float* S = sp + (long)b * CC * NN + j;
    for (int c = 0; c < CC; ++c) acc += S[(long)c * NN] * vsh[c];
    projected[(long)b * NN + perm[j]] = acc;
}

// serial LPAP hash-pool scan per batch (LDS tables), faithful to reference
__global__ void lpap_kernel(const float* ep, float* tgt, float* validf) {
    __shared__ float tv[CC];
    __shared__ int   td[CC];
    __shared__ int   tidv[CC];
    int b = blockIdx.x;
    int t = threadIdx.x;
    tv[t] = 0.f; td[t] = 0; tidv[t] = -1;
    __syncthreads();
    if (t == 0) {
        const float* E = ep + (long)b * NN;
        for (int i = 0; i < NN; ++i) {
            float cv = E[i];
            int cd = 0, cid = i;
            bool act = true;
            for (int k = 0; k < KEFF && act; ++k) {
                int pos = ((cid & 255) + cd) % CC;
                float etv = tv[pos]; int etd = td[pos]; int etid = tidv[pos];
                bool empty = etid < 0;
                bool steal = !empty && (etd < cd);
                bool place = empty || steal;
                if (place) { tv[pos] = cv; td[pos] = cd; tidv[pos] = cid; }
                if (steal) { cv = etv; cid = etid; cd = etd + 1; }
                else if (place) { act = false; }
                else { cd = cd + 1; }
            }
        }
    }
    __syncthreads();
    int id = tidv[t];
    bool valid = id >= 0;
    tgt[b * CC + t]    = (float)(valid ? id : 0);
    validf[b * CC + t] = valid ? 1.f : 0.f;
}

__global__ void weights_kernel(const float* ep, const float* validf, float* w) {
    int i = blockIdx.x * 256 + threadIdx.x;   // b*256+c
    int b = i >> 8, c = i & 255;
    float m = 0.f;
    for (int s = 0; s < STRIDE; ++s)
        m = fmaxf(m, fabsf(ep[(long)b * NN + s * CC + c]));
    w[i] = m * validf[i];
}

// ---------------- host-side transformer driver ----------------
struct TfParams {
    const float *embed_b, *embed_w;
    struct Layer {
        const float *b1, *b2, *bo, *bqkv, *ln1_b, *ln1_g, *ln2_b, *ln2_g,
                    *w1, *w2, *wo, *wqkv;
    } layer[LL];
    const float *lnf_b, *lnf_g, *out_b, *out_w, *pos;
};

struct TfW {   // pre-transposed bf16 weights (N x K layout, ldb = K)
    __bf16* embw;        // 256 x 32
    __bf16* wqkv[LL];    // 768 x 256
    __bf16* wo[LL];      // 256 x 256
    __bf16* w1[LL];      // 1024 x 256
    __bf16* w2[LL];      // 256 x 1024
    __bf16* outw;        // 8192 x 256
};

static TfParams make_params(void* const* d_in, int base) {
    auto f = [&](int i) { return (const float*)d_in[base + i]; };
    TfParams p;
    p.embed_b = f(0); p.embed_w = f(1);
    int i = 2;
    for (int l = 0; l < LL; ++l) {
        p.layer[l].b1 = f(i + 0);  p.layer[l].b2 = f(i + 1);
        p.layer[l].bo = f(i + 2);  p.layer[l].bqkv = f(i + 3);
        p.layer[l].ln1_b = f(i + 4); p.layer[l].ln1_g = f(i + 5);
        p.layer[l].ln2_b = f(i + 6); p.layer[l].ln2_g = f(i + 7);
        p.layer[l].w1 = f(i + 8);  p.layer[l].w2 = f(i + 9);
        p.layer[l].wo = f(i + 10); p.layer[l].wqkv = f(i + 11);
        i += 12;
    }
    p.lnf_b = f(26); p.lnf_g = f(27); p.out_b = f(28); p.out_w = f(29); p.pos = f(30);
    return p;
}

static void launch_gemm(hipStream_t s, const Gemm& g, int by = 1, int bz = 1) {
    long tiles = (long)(g.M / 16) * (g.N / 32);
    dim3 grid((unsigned)((tiles + 3) / 4), (unsigned)by, (unsigned)bz);
    if (g.bmode == 2)      gemm_wmma_kernel<2><<<grid, 128, 0, s>>>(g);
    else if (g.bmode == 1) gemm_wmma_kernel<1><<<grid, 128, 0, s>>>(g);
    else                   gemm_wmma_kernel<0><<<grid, 128, 0, s>>>(g);
}

static Gemm G(const float* A, long lda, const float* B, long ldb,
              const float* bias, const float* res, float* C, long ldc,
              int M, int K, int N, float scale = 1.0f, int act = 0) {
    Gemm g{};
    g.A = A; g.lda = lda; g.B = B; g.ldb = ldb; g.bmode = 0; g.Bbf = nullptr;
    g.bias = bias; g.residual = res;
    g.C = C; g.ldc = ldc;
    g.M = M; g.K = K; g.N = N; g.scale = scale; g.act = act;
    return g;
}
static Gemm GW(const float* A, long lda, const __bf16* Wt, long K,
               const float* bias, const float* res, float* C, long ldc,
               int M, int N, float scale = 1.0f, int act = 0) {
    Gemm g{};
    g.A = A; g.lda = lda; g.B = nullptr; g.ldb = K; g.bmode = 2; g.Bbf = Wt;
    g.bias = bias; g.residual = res;
    g.C = C; g.ldc = ldc;
    g.M = M; g.K = (int)K; g.N = N; g.scale = scale; g.act = act;
    return g;
}

static void prep_weights(hipStream_t st, const TfParams& p, int in_dim, const TfW& w) {
    transpose_cvt_kernel<<<(DD * 32) / 256, 256, 0, st>>>(p.embed_w, w.embw, in_dim, 32, DD);
    for (int l = 0; l < LL; ++l) {
        const TfParams::Layer& L = p.layer[l];
        transpose_cvt_kernel<<<(768 * 256) / 256, 256, 0, st>>>(L.wqkv, w.wqkv[l], 256, 256, 768);
        transpose_cvt_kernel<<<(256 * 256) / 256, 256, 0, st>>>(L.wo,   w.wo[l],   256, 256, 256);
        transpose_cvt_kernel<<<(1024 * 256) / 256, 256, 0, st>>>(L.w1,  w.w1[l],   256, 256, 1024);
        transpose_cvt_kernel<<<(256 * 1024) / 256, 256, 0, st>>>(L.w2,  w.w2[l],   1024, 1024, 256);
    }
    transpose_cvt_kernel<<<(8192 * 256) / 256, 256, 0, st>>>(p.out_w, w.outw, 256, 256, 8192);
}

// tokens: TOK x 32 (in_dim padded to 32)
static void run_transformer(hipStream_t st, const TfParams& p, const TfW& wt,
                            const float* tokens, float* logits_out,
                            float* xbuf, float* hbuf, float* qkvbuf,
                            float* obuf, float* mlpbuf, float* scorebuf) {
    const float inv_sqrt_hd = 0.17677669529663687f;   // 1/sqrt(32)

    // embed + bias (+pos)
    launch_gemm(st, GW(tokens, 32, wt.embw, 32, p.embed_b, nullptr, xbuf, DD, TOK, DD));
    add_pos_kernel<<<TOK * DD / 256, 256, 0, st>>>(xbuf, p.pos);

    for (int l = 0; l < LL; ++l) {
        const TfParams::Layer& L = p.layer[l];
        // ln1
        layernorm_kernel<<<TOK, 256, 0, st>>>(xbuf, L.ln1_g, L.ln1_b, hbuf);
        // qkv
        launch_gemm(st, GW(hbuf, DD, wt.wqkv[l], DD, L.bqkv, nullptr,
                           qkvbuf, 3 * DD, TOK, 3 * DD));
        // scores = Q @ K^T * inv_sqrt(hd), batched over (b,h); K rows contiguous
        {
            Gemm g = G(qkvbuf, 3 * DD, qkvbuf + DD, 3 * DD, nullptr, nullptr,
                       scorebuf, CC, CC, HD, CC, inv_sqrt_hd);
            g.bmode = 1;
            g.sAy = (long)CC * 3 * DD; g.sAz = HD;
            g.sBy = (long)CC * 3 * DD; g.sBz = HD;
            g.sCy = (long)HH * CC * CC; g.sCz = (long)CC * CC;
            launch_gemm(st, g, BB, HH);
        }
        softmax256_kernel<<<BB * HH * CC, 256, 0, st>>>(scorebuf);
        // O = A @ V, batched over (b,h) -> obuf[b*C+c, h*HD+d]
        {
            Gemm g = G(scorebuf, CC, qkvbuf + 2 * DD, 3 * DD, nullptr, nullptr,
                       obuf, DD, CC, CC, HD);
            g.sAy = (long)HH * CC * CC; g.sAz = (long)CC * CC;
            g.sBy = (long)CC * 3 * DD;  g.sBz = HD;
            g.sCy = (long)CC * DD;      g.sCz = HD;
            launch_gemm(st, g, BB, HH);
        }
        // x = x + O @ wo + bo
        launch_gemm(st, GW(obuf, DD, wt.wo[l], DD, L.bo, xbuf, xbuf, DD, TOK, DD));
        // ln2
        layernorm_kernel<<<TOK, 256, 0, st>>>(xbuf, L.ln2_g, L.ln2_b, hbuf);
        // mlp: gelu(h@w1+b1) @ w2 + b2 + x
        launch_gemm(st, GW(hbuf, DD, wt.w1[l], DD, L.b1, nullptr,
                           mlpbuf, 4 * DD, TOK, 4 * DD, 1.0f, /*gelu*/1));
        launch_gemm(st, GW(mlpbuf, 4 * DD, wt.w2[l], 4 * DD, L.b2, xbuf, xbuf, DD,
                           TOK, DD));
    }
    // final LN + output projection
    layernorm_kernel<<<TOK, 256, 0, st>>>(xbuf, p.lnf_g, p.lnf_b, hbuf);
    launch_gemm(st, GW(hbuf, DD, wt.outw, DD, p.out_b, nullptr, logits_out, NN, TOK, NN));
}

// ---------------- entry ----------------
extern "C" void kernel_launch(void* const* d_in, const int* in_sizes, int n_in,
                              void* d_out, int out_size, void* d_ws, size_t ws_size,
                              hipStream_t stream) {
    (void)in_sizes; (void)n_in; (void)out_size; (void)ws_size;

    const float* learned_energy = (const float*)d_in[0];
    const int*   perm           = (const int*)d_in[1];
    TfParams sur = make_params(d_in, 2);
    TfParams dec = make_params(d_in, 33);
    const float* slt = (const float*)d_in[64];

    // output layout (flat f32, reference return order)
    float* out = (float*)d_out;
    long off = 0;
    float* o_ep         = out + off; off += (long)BB * NN;          // ep
    float* o_sur_logits = out + off; off += (long)BB * CC * NN;     // sur_logits
    float* o_dec_tokens = out + off; off += (long)BB * CC * 3;      // dec_tokens
    float* o_dec_logits = out + off; off += (long)BB * CC * NN;     // dec_logits
    float* o_projected  = out + off; off += (long)BB * NN;          // projected
    float* o_sp         = out + off; off += (long)BB * CC * NN;     // sp
    float* o_doubt      = out + off; off += (long)BB * CC;          // doubt
    float* o_support    = out + off; off += (long)BB * CC;          // support
    float* o_T          = out + off; off += 1;                      // T
    float* o_tidx       = out + off; off += (long)BB * CC;          // target_idx
    float* o_valid      = out + off; off += (long)BB * CC;          // valid
    float* o_weights    = out + off; off += (long)BB * CC;          // weights

    // workspace layout: f32 region then bf16 region (16B-aligned chunks)
    float* w = (float*)d_ws;
    float* tokens   = w;                               // 2048*32
    float* xbuf     = tokens + (long)TOK * 32;         // 2048*256
    float* hbuf     = xbuf + (long)TOK * DD;
    float* qkvbuf   = hbuf + (long)TOK * DD;           // 2048*768
    float* obuf     = qkvbuf + (long)TOK * 3 * DD;     // 2048*256
    float* mlpbuf   = obuf + (long)TOK * DD;           // 2048*1024
    float* scorebuf = mlpbuf + (long)TOK * 4 * DD;     // 64*256*256
    __bf16* bw = (__bf16*)(scorebuf + (long)BB * HH * CC * CC);

    TfW wt;
    wt.embw = bw;                       bw += (long)DD * 32;
    for (int l = 0; l < LL; ++l) {
        wt.wqkv[l] = bw;                bw += (long)768 * 256;
        wt.wo[l]   = bw;                bw += (long)256 * 256;
        wt.w1[l]   = bw;                bw += (long)1024 * 256;
        wt.w2[l]   = bw;                bw += (long)256 * 1024;
    }
    wt.outw = bw;                       bw += (long)8192 * 256;

    // 1) ep = e[:, perm]
    permute_kernel<<<dim3(NN / 256, BB), 256, 0, stream>>>(learned_energy, perm, o_ep);
    // 2) tokens[b,c,s] = ep[b, s*C + c]
    tokens_kernel<<<TOK, STRIDE, 0, stream>>>(o_ep, tokens);
    // 3) surrogate transformer -> sur_logits
    prep_weights(stream, sur, STRIDE, wt);
    run_transformer(stream, sur, wt, tokens, o_sur_logits,
                    xbuf, hbuf, qkvbuf, obuf, mlpbuf, scorebuf);
    // 4) softmax moments -> dec_tokens (val, dexp, ent)
    sur_post_kernel<<<TOK, 256, 0, stream>>>(o_sur_logits, o_ep, o_dec_tokens);
    // 5) decoder transformer -> dec_logits (pad K=3 -> 32 so GEMM stays branch-free)
    pad_dec_tokens_kernel<<<TOK, 32, 0, stream>>>(o_dec_tokens, tokens);
    prep_weights(stream, dec, 3, wt);
    run_transformer(stream, dec, wt, tokens, o_dec_logits,
                    xbuf, hbuf, qkvbuf, obuf, mlpbuf, scorebuf);
    // 6) T
    tval_kernel<<<1, 1, 0, stream>>>(slt, o_T);
    // 7) sp / doubt / support
    dec_post_kernel<<<TOK, 256, 0, stream>>>(o_dec_logits, slt, o_sp, o_doubt, o_support);
    // 8) projected (perm is a permutation -> scatter covers all of N)
    projected_kernel<<<dim3(NN / 256, BB), 256, 0, stream>>>(o_sp, o_dec_tokens, perm, o_projected);
    // 9) LPAP pool -> target_idx, valid
    lpap_kernel<<<BB, CC, 0, stream>>>(o_ep, o_tidx, o_valid);
    // 10) weights = max_s |ep| * valid
    weights_kernel<<<BB * CC / 256, 256, 0, stream>>>(o_ep, o_valid, o_weights);
}
